// CausalSelfAttention_3530463117417
// MI455X (gfx1250) — matrix operand order repaired and checked by reading, not compiled, not run on hardware
//
#include <hip/hip_runtime.h>
#include <hip/hip_bf16.h>

// ---- problem constants (from reference) ----
#define BB    2
#define TSEQ  2048
#define CEMB  1024
#define NH    16
#define NKV   4
#define HD    64
#define BT    (BB * TSEQ)           // 4096 rows
#define NQKV  (CEMB + 2 * NKV * HD) // 1536 = q(1024)|k(256)|v(256)

typedef __bf16 bf16_t;
typedef bf16_t  v16bf __attribute__((ext_vector_type(16)));
typedef float   v8f   __attribute__((ext_vector_type(8)));
typedef unsigned v4u  __attribute__((ext_vector_type(4)));
typedef int      v4i  __attribute__((ext_vector_type(4)));
typedef int      v8i  __attribute__((ext_vector_type(8)));

struct __align__(16) U4 { unsigned x, y, z, w; };
union Frag16 { v16bf v; U4 q[2]; };

__device__ inline unsigned short f2bf(float f) {
    union { float f; unsigned u; } t; t.f = f;
    unsigned u = t.u;
    u += 0x7FFFu + ((u >> 16) & 1u);   // round-to-nearest-even
    return (unsigned short)(u >> 16);
}

__device__ inline v8f vzero8() {
    v8f v;
    for (int i = 0; i < 8; ++i) v[i] = 0.0f;
    return v;
}

// ---------------- elementwise converts ----------------
__global__ void f32_to_bf16_kernel(const float* __restrict__ in,
                                   unsigned short* __restrict__ out, int n) {
    int i = blockIdx.x * blockDim.x + threadIdx.x;
    if (i < n) out[i] = f2bf(in[i]);
}

// transpose + convert: in[R][C] f32 -> out[C][R] bf16  (K-contiguous weights)
__global__ void f32_to_bf16_T_kernel(const float* __restrict__ in,
                                     unsigned short* __restrict__ out,
                                     int R, int C) {
    int i = blockIdx.x * blockDim.x + threadIdx.x;
    if (i >= R * C) return;
    int r = i / C, c = i % C;
    out[(size_t)c * R + r] = f2bf(in[i]);
}

// fused q|k|v column-concat, stored TRANSPOSED: out[n][k], n in [0,1536), k in [0,1024)
__global__ void pack_wqkv_T_kernel(const float* __restrict__ Wq,
                                   const float* __restrict__ Wk,
                                   const float* __restrict__ Wv,
                                   unsigned short* __restrict__ out) {
    int i = blockIdx.x * blockDim.x + threadIdx.x;   // over 1024*1536
    if (i >= CEMB * NQKV) return;
    int k = i / NQKV, n = i % NQKV;
    float v;
    if (n < 1024)       v = Wq[k * 1024 + n];
    else if (n < 1280)  v = Wk[k * 256 + (n - 1024)];
    else                v = Wv[k * 256 + (n - 1280)];
    out[(size_t)n * CEMB + k] = f2bf(v);
}

// ---------------- TDM 2D tile load: global (row-major, K-contig) -> LDS ------
// Pads 4 DWORDs after every 16 DWORDs (one 32-elem bf16 row) -> 40-elem LDS rows.
#if __has_builtin(__builtin_amdgcn_tensor_load_to_lds)
#define HAVE_TDM 1
__device__ inline void tdm_load_tile(unsigned ldsOff, const unsigned short* gptr,
                                     unsigned tile_rows, unsigned stride_elems) {
    unsigned long long ga = (unsigned long long)(uintptr_t)gptr;
    v4u g0;
    g0[0] = 1u;                                   // count=1, user descriptor
    g0[1] = ldsOff;                               // lds_addr (bytes)
    g0[2] = (unsigned)(ga & 0xFFFFFFFFu);         // global_addr[31:0]
    g0[3] = (unsigned)((ga >> 32) & 0x01FFFFFFu)  // global_addr[56:32]
            | 0x80000000u;                        // type=2 ("image")
    v8i g1;
    g1[0] = (int)((1u << 16)      // data_size = 2 bytes
                | (1u << 20)      // pad_enable
                | (3u << 22)      // pad_interval: every 16 DWORDs
                | (3u << 25));    // pad_amount: 4 DWORDs
    g1[1] = (int)((stride_elems & 0xFFFFu) << 16);          // tensor_dim0[15:0]
    g1[2] = (int)(((stride_elems >> 16) & 0xFFFFu)          // tensor_dim0[31:16]
                | ((tile_rows & 0xFFFFu) << 16));           // tensor_dim1[15:0]
    g1[3] = (int)(32u << 16);                               // tile_dim0 = 32
    g1[4] = (int)(tile_rows & 0xFFFFu);                     // tile_dim1
    g1[5] = (int)stride_elems;                              // tensor_dim0_stride[31:0]
    g1[6] = 0;
    g1[7] = 0;
    v4i gz = {0, 0, 0, 0};
#if defined(__clang_major__) && (__clang_major__ >= 23)
    v8i gz8 = {0, 0, 0, 0, 0, 0, 0, 0};
    __builtin_amdgcn_tensor_load_to_lds(g0, g1, gz, gz, gz8, 0);
#else
    __builtin_amdgcn_tensor_load_to_lds(g0, g1, gz, gz, 0);
#endif
}
#else
#define HAVE_TDM 0
#endif

// ---------------- tiled WMMA GEMM: C[M,N] f32 = A[M,K] @ Bt[N,K] (bf16) ------
// block tile 64(M) x 128(N), 8 waves (4x2), each wave 16x64 (4 WMMA n-tiles).
// TDM double-buffers both tiles into LDS; waves compute on the other buffer.
__global__ __launch_bounds__(256) void gemm_bf16_f32(
        const unsigned short* __restrict__ A,
        const unsigned short* __restrict__ Bt,
        float* __restrict__ C, int M, int N, int K) {
    __shared__ __align__(16) unsigned short AsBuf[2][64 * 40];   // [m][k] stride 40
    __shared__ __align__(16) unsigned short BtBuf[2][128 * 40];  // [n][k] stride 40

    const int tid  = threadIdx.x;
    const int lane = tid & 31;
    const int wave = tid >> 5;
    const int mb0  = blockIdx.y * 64;
    const int nb0  = blockIdx.x * 128;
    const int mw   = (wave >> 1) * 16;
    const int nw   = (wave & 1) * 64;
    const int sel  = (lane >> 4) & 1;
    const int l15  = lane & 15;

    v8f acc[4];
    for (int j = 0; j < 4; ++j) acc[j] = vzero8();

#if HAVE_TDM
    unsigned asOff[2], btOff[2];
    for (int b = 0; b < 2; ++b) {
        asOff[b] = (unsigned)(uintptr_t)&AsBuf[b][0];
        btOff[b] = (unsigned)(uintptr_t)&BtBuf[b][0];
    }
    if (wave == 0) {   // TDM ignores EXEC; issue once per workgroup (wave 0)
        tdm_load_tile(asOff[0], A  + (size_t)mb0 * K, 64,  (unsigned)K);
        tdm_load_tile(btOff[0], Bt + (size_t)nb0 * K, 128, (unsigned)K);
    }
#endif

    int cur = 0;
    for (int kt = 0; kt < K; kt += 32) {
#if HAVE_TDM
        int nxt = cur ^ 1;
        if (wave == 0) {
            if (kt + 32 < K) {
                tdm_load_tile(asOff[nxt], A  + (size_t)mb0 * K + kt + 32, 64,  (unsigned)K);
                tdm_load_tile(btOff[nxt], Bt + (size_t)nb0 * K + kt + 32, 128, (unsigned)K);
                __builtin_amdgcn_s_wait_tensorcnt((short)2);  // current 2 ops done
            } else {
                __builtin_amdgcn_s_wait_tensorcnt((short)0);
            }
        }
        __syncthreads();
#else
        // fallback: manual cooperative staging (both operands K-contiguous)
        {
            U4 av = *(const U4*)(A + (size_t)(mb0 + (tid >> 2)) * K + kt + (tid & 3) * 8);
            *(U4*)&AsBuf[cur][(tid >> 2) * 40 + (tid & 3) * 8] = av;
            for (int i = 0; i < 2; ++i) {
                int v  = tid + i * 256;     // 512 vec8 slots for 128x32
                int n  = v >> 2;
                int kc = (v & 3) * 8;
                U4 bv = *(const U4*)(Bt + (size_t)(nb0 + n) * K + kt + kc);
                *(U4*)&BtBuf[cur][n * 40 + kc] = bv;
            }
        }
        __syncthreads();
#endif
        const unsigned short* As = AsBuf[cur];
        const unsigned short* Bs = BtBuf[cur];

        // A frag: lane<16 -> M=lane, K 0..7|16..23 ; lane>=16 -> K 8..15|24..31
        Frag16 af;
        {
            int row = mw + l15;
            int off = sel * 8;
            af.q[0] = *(const U4*)&As[row * 40 + off];
            af.q[1] = *(const U4*)&As[row * 40 + 16 + off];
        }
        // B frags: lane<16 -> N=lane, K 0..7|16..23 ; lane>=16 -> K 8..15|24..31
        for (int j = 0; j < 4; ++j) {
            Frag16 bf;
            int col = nw + j * 16 + l15;
            int ko  = sel * 8;
            bf.q[0] = *(const U4*)&Bs[col * 40 + ko];
            bf.q[1] = *(const U4*)&Bs[col * 40 + ko + 16];
            acc[j] = __builtin_amdgcn_wmma_f32_16x16x32_bf16(
                         false, af.v, false, bf.v, (short)0, acc[j], false, false);
        }
        __syncthreads();
        cur ^= 1;
    }

    for (int j = 0; j < 4; ++j)
        for (int r = 0; r < 8; ++r) {
            int row = mb0 + mw + r + 8 * sel;
            int col = nb0 + nw + j * 16 + l15;
            C[(size_t)row * N + col] = acc[j][r];
        }
}

// ---------------- RoPE + layout split ----------------
// qkvf[bt][1536] f32 -> qb [B,H,T,D] bf16 (rope), kb [B,Hkv,T,D] bf16 (rope),
// vt [B,Hkv,D,T] bf16 (transposed for the P@V B-fragment)
__global__ void rope_split_kernel(const float* __restrict__ qkvf,
                                  const float* __restrict__ cosp,
                                  const float* __restrict__ sinp,
                                  unsigned short* __restrict__ qb,
                                  unsigned short* __restrict__ kb,
                                  unsigned short* __restrict__ vt) {
    int idx = blockIdx.x * blockDim.x + threadIdx.x;
    if (idx >= BT * NQKV) return;
    int col = idx % NQKV;
    int bt  = idx / NQKV;
    int t   = bt % TSEQ;
    int b   = bt / TSEQ;
    const float* row = qkvf + (size_t)bt * NQKV;

    if (col < 1024) {
        int h = col >> 6, d = col & 63;
        float out;
        if (d < 32) {
            float x1 = row[col], x2 = row[col + 32];
            out = x1 * cosp[t * 32 + d] + x2 * sinp[t * 32 + d];
        } else {
            int dd = d - 32;
            float x1 = row[col - 32], x2 = row[col];
            out = -x1 * sinp[t * 32 + dd] + x2 * cosp[t * 32 + dd];
        }
        qb[(((size_t)b * NH + h) * TSEQ + t) * HD + d] = f2bf(out);
    } else if (col < 1280) {
        int cc = col - 1024;
        int h = cc >> 6, d = cc & 63;
        float out;
        if (d < 32) {
            float x1 = row[col], x2 = row[col + 32];
            out = x1 * cosp[t * 32 + d] + x2 * sinp[t * 32 + d];
        } else {
            int dd = d - 32;
            float x1 = row[col - 32], x2 = row[col];
            out = -x1 * sinp[t * 32 + dd] + x2 * cosp[t * 32 + dd];
        }
        kb[(((size_t)b * NKV + h) * TSEQ + t) * HD + d] = f2bf(out);
    } else {
        int cc = col - 1280;
        int h = cc >> 6, d = cc & 63;
        vt[(((size_t)b * NKV + h) * HD + d) * TSEQ + t] = f2bf(row[col]);
    }
}

// ---------------- flash attention (causal, GQA rep=4) ----------------
// 1 wave = one 16-row Q tile; 4 waves/block = 64 Q rows of one (b,h).
__global__ __launch_bounds__(128) void flash_attn_kernel(
        const unsigned short* __restrict__ qb,
        const unsigned short* __restrict__ kb,
        const unsigned short* __restrict__ vt,
        unsigned short* __restrict__ yb) {
    __shared__ __align__(16) unsigned short pstage[4][16 * 40]; // per-wave P tile

    const int lane = threadIdx.x & 31;
    const int wave = threadIdx.x >> 5;
    const int nqb  = TSEQ / 64;
    int bid  = blockIdx.x;
    int qblk = bid % nqb;
    int h    = (bid / nqb) % NH;
    int b    = bid / (nqb * NH);
    int hkv  = h >> 2;
    int m0   = qblk * 64 + wave * 16;
    const int sel = (lane >> 4) & 1;
    const int l15 = lane & 15;

    const unsigned short* Qp = qb + (((size_t)b * NH  + h)   * TSEQ) * HD;
    const unsigned short* Kp = kb + (((size_t)b * NKV + hkv) * TSEQ) * HD;
    const unsigned short* Vp = vt + (((size_t)b * NKV + hkv) * HD)   * TSEQ;

    // Q A-fragments (16x32 each), two K-chunks covering D=64
    Frag16 qf[2];
    {
        int row = m0 + l15;
        for (int c = 0; c < 2; ++c) {
            int o0 = 32 * c + sel * 8;
            qf[c].q[0] = *(const U4*)(Qp + (size_t)row * HD + o0);
            qf[c].q[1] = *(const U4*)(Qp + (size_t)row * HD + o0 + 16);
        }
    }

    v8f o[4];
    for (int j = 0; j < 4; ++j) o[j] = vzero8();
    float mi[8], li[8];
    for (int r = 0; r < 8; ++r) { mi[r] = -1e30f; li[r] = 0.0f; }

    unsigned short* ps = pstage[wave];

    for (int k0 = 0; k0 < m0 + 16; k0 += 32) {
        float s[2][8];
        for (int n = 0; n < 2; ++n) {
            int nb = n * 16;
            v8f a = vzero8();
            for (int c = 0; c < 2; ++c) {
                Frag16 kf;                       // B = K^T chunk (32d x 16keys)
                int key = k0 + nb + l15;
                if (key >= TSEQ) key = TSEQ - 1; // masked anyway
                int d0 = 32 * c + sel * 8;
                kf.q[0] = *(const U4*)(Kp + (size_t)key * HD + d0);
                kf.q[1] = *(const U4*)(Kp + (size_t)key * HD + d0 + 16);
                a = __builtin_amdgcn_wmma_f32_16x16x32_bf16(
                        false, qf[c].v, false, kf.v, (short)0, a, false, false);
            }
            for (int r = 0; r < 8; ++r) {
                int R  = r + 8 * sel;            // row within tile
                int kc = k0 + nb + l15;          // key column
                float v = a[r] * 0.125f;         // 1/sqrt(64)
                if (kc > m0 + R) v = -1e30f;     // causal mask
                s[n][r] = v;
            }
        }
        // online softmax (rows live in 16-lane halves; xor<=8 stays in half)
        float corr[8];
        for (int r = 0; r < 8; ++r) {
            float mx = fmaxf(s[0][r], s[1][r]);
            for (int off = 8; off > 0; off >>= 1)
                mx = fmaxf(mx, __shfl_xor(mx, off, 32));
            float mnew = fmaxf(mi[r], mx);
            float cr = __expf(mi[r] - mnew);
            float p0 = __expf(s[0][r] - mnew);
            float p1 = __expf(s[1][r] - mnew);
            float rs = p0 + p1;
            for (int off = 8; off > 0; off >>= 1)
                rs += __shfl_xor(rs, off, 32);
            li[r] = li[r] * cr + rs;
            mi[r] = mnew;
            corr[r] = cr;
            s[0][r] = p0;
            s[1][r] = p1;
        }
        for (int j = 0; j < 4; ++j)
            for (int r = 0; r < 8; ++r) o[j][r] *= corr[r];

        // stage P (C-layout -> LDS -> A-layout), stride 40 = conflict-free
        for (int n = 0; n < 2; ++n)
            for (int r = 0; r < 8; ++r)
                ps[(r + 8 * sel) * 40 + n * 16 + l15] = f2bf(s[n][r]);
        asm volatile("s_wait_dscnt 0" ::: "memory");
        Frag16 pf;
        pf.q[0] = *(const U4*)&ps[l15 * 40 + sel * 16];
        pf.q[1] = *(const U4*)&ps[l15 * 40 + sel * 16 + 8];

        // O += P @ V  (V transposed: B-frag = 16 contiguous keys per lane)
        int koff = k0 + sel * 16;
        if (koff > TSEQ - 16) koff = TSEQ - 16;  // p==0 there anyway
        for (int j = 0; j < 4; ++j) {
            Frag16 vf;
            const unsigned short* vp = Vp + (size_t)(j * 16 + l15) * TSEQ + koff;
            vf.q[0] = *(const U4*)(vp);
            vf.q[1] = *(const U4*)(vp + 8);
            o[j] = __builtin_amdgcn_wmma_f32_16x16x32_bf16(
                       false, pf.v, false, vf.v, (short)0, o[j], false, false);
        }
        asm volatile("s_wait_dscnt 0" ::: "memory");
    }

    float inv[8];
    for (int r = 0; r < 8; ++r) inv[r] = 1.0f / li[r];
    for (int j = 0; j < 4; ++j)
        for (int r = 0; r < 8; ++r) {
            int row = m0 + r + 8 * sel;
            int col = h * HD + j * 16 + l15;
            yb[((size_t)b * TSEQ + row) * CEMB + col] = f2bf(o[j][r] * inv[r]);
        }
}

// ---------------- launcher ----------------
extern "C" void kernel_launch(void* const* d_in, const int* in_sizes, int n_in,
                              void* d_out, int out_size, void* d_ws, size_t ws_size,
                              hipStream_t stream) {
    const float* x    = (const float*)d_in[0];
    const float* cosp = (const float*)d_in[1];
    const float* sinp = (const float*)d_in[2];
    const float* Wq   = (const float*)d_in[3];
    const float* Wk   = (const float*)d_in[4];
    const float* Wv   = (const float*)d_in[5];
    const float* Wo   = (const float*)d_in[6];
    float* out = (float*)d_out;

    char* ws = (char*)d_ws;
    size_t off = 0;
    auto alloc = [&](size_t bytes) -> void* {
        void* p = ws + off;
        off = (off + bytes + 255) & ~(size_t)255;
        return p;
    };

    unsigned short* xb    = (unsigned short*)alloc((size_t)BT * CEMB * 2);            // 8 MB
    unsigned short* wqkvT = (unsigned short*)alloc((size_t)NQKV * CEMB * 2);          // 3 MB
    unsigned short* woT   = (unsigned short*)alloc((size_t)CEMB * CEMB * 2);          // 2 MB
    float*          qkvf  = (float*)         alloc((size_t)BT * NQKV * 4);            // 24 MB
    unsigned short* qb    = (unsigned short*)alloc((size_t)BB * NH  * TSEQ * HD * 2); // 8 MB
    unsigned short* kb    = (unsigned short*)alloc((size_t)BB * NKV * TSEQ * HD * 2); // 2 MB
    unsigned short* vt    = (unsigned short*)alloc((size_t)BB * NKV * HD * TSEQ * 2); // 2 MB
    unsigned short* yb    = (unsigned short*)alloc((size_t)BT * CEMB * 2);            // 8 MB

    // 1) converts (weights transposed so GEMM operands are K-contiguous for TDM)
    f32_to_bf16_kernel<<<(BT * CEMB) / 256, 256, 0, stream>>>(x, xb, BT * CEMB);
    pack_wqkv_T_kernel<<<(CEMB * NQKV) / 256, 256, 0, stream>>>(Wq, Wk, Wv, wqkvT);
    f32_to_bf16_T_kernel<<<(CEMB * CEMB) / 256, 256, 0, stream>>>(Wo, woT, CEMB, CEMB);

    // 2) fused QKV projection: [4096,1536] = [4096,1024] @ [1024,1536]
    gemm_bf16_f32<<<dim3(NQKV / 128, BT / 64), 256, 0, stream>>>(
        xb, wqkvT, qkvf, BT, NQKV, CEMB);

    // 3) RoPE + split/convert
    rope_split_kernel<<<(BT * NQKV) / 256, 256, 0, stream>>>(
        qkvf, cosp, sinp, qb, kb, vt);

    // 4) causal flash attention
    flash_attn_kernel<<<BB * NH * (TSEQ / 64), 128, 0, stream>>>(qb, kb, vt, yb);

    // 5) output projection: [4096,1024] = [4096,1024] @ [1024,1024] -> f32 out
    gemm_bf16_f32<<<dim3(CEMB / 128, BT / 64), 256, 0, stream>>>(
        yb, woT, out, BT, CEMB, CEMB);
}
